// GINGraphModel_11665131176543
// MI455X (gfx1250) — matrix operand order repaired and checked by reading, not compiled, not run on hardware
//
#include <hip/hip_runtime.h>
#include <hip/hip_bf16.h>

typedef __attribute__((ext_vector_type(16))) _Float16 v16h;
typedef __attribute__((ext_vector_type(8)))  float    v8f;

// ---------------------------------------------------------------------------
// Simple float4 grid-stride copy: agg initialization (agg = x term of GIN).
// ---------------------------------------------------------------------------
__global__ void copy_f32_kernel(const float* __restrict__ src,
                                float* __restrict__ dst, size_t n4) {
  size_t i = (size_t)blockIdx.x * blockDim.x + threadIdx.x;
  size_t stride = (size_t)gridDim.x * blockDim.x;
  const float4* s4 = (const float4*)src;
  float4* d4 = (float4*)dst;
  for (; i < n4; i += stride) d4[i] = s4[i];
}

// ---------------------------------------------------------------------------
// Edge-parallel scatter-add: one wave32 per edge, lane owns 4 features.
// Gather of x[src] is a coalesced 512B float4 load per wave; adds are
// global_atomic_add_f32 which resolve in L2 (entire node array is L2-resident
// on MI455X: 192MB L2 >> 51MB per feature matrix).
// ---------------------------------------------------------------------------
__global__ void scatter_add_kernel(const float* __restrict__ x,
                                   float* __restrict__ agg,
                                   const int* __restrict__ src,
                                   const int* __restrict__ dst,
                                   int E, int D) {
  int wave  = (int)(((size_t)blockIdx.x * blockDim.x + threadIdx.x) >> 5);
  int lane  = threadIdx.x & 31;
  int nwav  = (int)(((size_t)gridDim.x * blockDim.x) >> 5);
  int chunks = D >> 7;  // D/128: features handled in 128-wide passes (32 lanes x float4)
  for (int e = wave; e < E; e += nwav) {
    int s = src[e];
    int d = dst[e];
    const float4* xs = (const float4*)(x + (size_t)s * D);
    float* ag = agg + (size_t)d * D;
    for (int c = 0; c < chunks; ++c) {
      int idx4 = c * 32 + lane;           // float4 index within row
      float4 v = xs[idx4];
      int f = idx4 * 4;
      atomicAdd(ag + f + 0, v.x);
      atomicAdd(ag + f + 1, v.y);
      atomicAdd(ag + f + 2, v.z);
      atomicAdd(ag + f + 3, v.w);
    }
  }
}

// ---------------------------------------------------------------------------
// Weight prep: W (K x Nout, f32 row-major)  ->  WT (Nout x K, f16 row-major).
// Transposed+f16 so that each lane's WMMA B-fragment is one contiguous 32B
// load (v16h) along K.
// ---------------------------------------------------------------------------
__global__ void w_to_f16T_kernel(const float* __restrict__ W,
                                 _Float16* __restrict__ WT, int K, int Nout) {
  int n = blockIdx.x * blockDim.x + threadIdx.x;
  int k = blockIdx.y * blockDim.y + threadIdx.y;
  if (n < Nout && k < K)
    WT[(size_t)n * K + k] = (_Float16)W[(size_t)k * Nout + n];
}

// ---------------------------------------------------------------------------
// WMMA GEMM: Out[M x NOUT] = act(A[M x K] @ W[K x NOUT] + bias)
//  - v_wmma_f32_16x16x32_f16, f32 accumulate
//  - one wave per 16-row stripe; A fragments for the FULL K dimension are
//    held in registers (<=64 VGPRs) and reused across every N-tile, so the
//    activation matrix is read from memory exactly once.
//  - B fragments: lane (lane&15)=column, (lane>>4) selects K 0-15 / 16-31 of
//    the 32-wide K step  ->  contiguous v16h load from transposed f16 weights.
// ---------------------------------------------------------------------------
template <int K, int NOUT, bool RELU>
__global__ void gemm_bias_act_kernel(const float* __restrict__ A,
                                     const _Float16* __restrict__ WT,
                                     const float* __restrict__ bias,
                                     float* __restrict__ Out, int M) {
  constexpr int KSTEPS = K / 32;
  constexpr int NTILES = NOUT / 16;

  int wave  = threadIdx.x >> 5;
  int lane  = threadIdx.x & 31;
  int mtile = blockIdx.x * (blockDim.x >> 5) + wave;
  if (mtile * 16 >= M) return;  // wave-uniform: EXEC stays all-ones for WMMA

  int half = lane >> 4;   // 0: lanes 0-15, 1: lanes 16-31
  int mrow = lane & 15;
  int row  = mtile * 16 + mrow;
  const float* ap = A + (size_t)row * K;

  // A fragments (16x32 f16 layout): lane<16 holds K {k0..k0+15},
  // lane>=16 holds K {k0+16..k0+31}.
  v16h afrag[KSTEPS];
#pragma unroll
  for (int ks = 0; ks < KSTEPS; ++ks) {
    int k0  = ks * 32;
    int kb0 = k0 + half * 16;
    int kb1 = k0 + half * 16 + 8;
#pragma unroll
    for (int j = 0; j < 8; ++j) {
      afrag[ks][j]     = (_Float16)ap[kb0 + j];
      afrag[ks][8 + j] = (_Float16)ap[kb1 + j];
    }
  }

#pragma unroll
  for (int nt = 0; nt < NTILES; ++nt) {
    int gcol = nt * 16 + mrow;
    const _Float16* wp = WT + (size_t)gcol * K + half * 16;
    v8f c = {};
#pragma unroll
    for (int ks = 0; ks < KSTEPS; ++ks) {
      v16h b = *(const v16h*)(wp + ks * 32);
      c = __builtin_amdgcn_wmma_f32_16x16x32_f16(
          /*neg_a=*/false, afrag[ks], /*neg_b=*/false, b,
          /*c_mod=*/(short)0, c, /*reuse_a=*/false, /*reuse_b=*/false);
    }
    float bb = bias[gcol];
#pragma unroll
    for (int r = 0; r < 8; ++r) {
      float v = c[r] + bb;
      if (RELU) v = fmaxf(v, 0.0f);
      // C/D layout: VGPR r -> row (8*half + r), column = lane&15.
      Out[(size_t)(mtile * 16 + half * 8 + r) * NOUT + gcol] = v;
    }
  }
}

// ---------------------------------------------------------------------------
// Host-side orchestration
// ---------------------------------------------------------------------------
static inline void* ws_alloc(char* base, size_t& off, size_t bytes) {
  void* p = base + off;
  off += (bytes + 255) & ~(size_t)255;
  return p;
}

extern "C" void kernel_launch(void* const* d_in, const int* in_sizes, int n_in,
                              void* d_out, int out_size, void* d_ws, size_t ws_size,
                              hipStream_t stream) {
  const float* x   = (const float*)d_in[0];
  const int*   ei  = (const int*)d_in[1];
  const float* w1a = (const float*)d_in[2];
  const float* b1a = (const float*)d_in[3];
  const float* w1b = (const float*)d_in[4];
  const float* b1b = (const float*)d_in[5];
  const float* w2a = (const float*)d_in[6];
  const float* b2a = (const float*)d_in[7];
  const float* w2b = (const float*)d_in[8];
  const float* b2b = (const float*)d_in[9];
  const float* w3a = (const float*)d_in[10];
  const float* b3a = (const float*)d_in[11];
  const float* w3b = (const float*)d_in[12];
  const float* b3b = (const float*)d_in[13];

  const int N = in_sizes[0] / 128;   // 50000 nodes
  const int E = in_sizes[1] / 2;     // 800000 edges
  const int* src = ei;
  const int* dst = ei + E;

  // Workspace: 3 rotating N x 256 f32 buffers (~154MB) + tiny f16 weights.
  char* ws = (char*)d_ws;
  size_t off = 0;
  float* bufA = (float*)ws_alloc(ws, off, (size_t)N * 256 * sizeof(float));
  float* bufB = (float*)ws_alloc(ws, off, (size_t)N * 256 * sizeof(float));
  float* bufC = (float*)ws_alloc(ws, off, (size_t)N * 256 * sizeof(float));
  _Float16* w1aT = (_Float16*)ws_alloc(ws, off, 128 * 128 * sizeof(_Float16));
  _Float16* w1bT = (_Float16*)ws_alloc(ws, off, 256 * 128 * sizeof(_Float16));
  _Float16* w2aT = (_Float16*)ws_alloc(ws, off, 256 * 256 * sizeof(_Float16));
  _Float16* w2bT = (_Float16*)ws_alloc(ws, off, 128 * 256 * sizeof(_Float16));
  _Float16* w3aT = (_Float16*)ws_alloc(ws, off, 128 * 128 * sizeof(_Float16));
  _Float16* w3bT = (_Float16*)ws_alloc(ws, off, 128 * 128 * sizeof(_Float16));

  // --- weight prep (tiny) ---
  dim3 tb(16, 16);
  w_to_f16T_kernel<<<dim3(128 / 16, 128 / 16), tb, 0, stream>>>(w1a, w1aT, 128, 128);
  w_to_f16T_kernel<<<dim3(256 / 16, 128 / 16), tb, 0, stream>>>(w1b, w1bT, 128, 256);
  w_to_f16T_kernel<<<dim3(256 / 16, 256 / 16), tb, 0, stream>>>(w2a, w2aT, 256, 256);
  w_to_f16T_kernel<<<dim3(128 / 16, 256 / 16), tb, 0, stream>>>(w2b, w2bT, 256, 128);
  w_to_f16T_kernel<<<dim3(128 / 16, 128 / 16), tb, 0, stream>>>(w3a, w3aT, 128, 128);
  w_to_f16T_kernel<<<dim3(128 / 16, 128 / 16), tb, 0, stream>>>(w3b, w3bT, 128, 128);

  const int mtiles  = N / 16;                 // 3125, N % 16 == 0
  const int gblocks = (mtiles + 7) / 8;       // 8 waves (16x16 tiles) per block
  const int eblocks = (E + 7) / 8;            // 1 wave per edge
  const int cblocks128 = ((size_t)N * 128 / 4 + 255) / 256;
  const int cblocks256 = ((size_t)N * 256 / 4 + 255) / 256;

  // ---------------- Layer 1: agg = x + scatter(x); MLP 128->128->256 --------
  copy_f32_kernel<<<cblocks128, 256, 0, stream>>>(x, bufA, (size_t)N * 128 / 4);
  scatter_add_kernel<<<eblocks, 256, 0, stream>>>(x, bufA, src, dst, E, 128);
  gemm_bias_act_kernel<128, 128, true><<<gblocks, 256, 0, stream>>>(bufA, w1aT, b1a, bufB, N);
  gemm_bias_act_kernel<128, 256, true><<<gblocks, 256, 0, stream>>>(bufB, w1bT, b1b, bufC, N);

  // ---------------- Layer 2: agg = h + scatter(h); MLP 256->256->128 --------
  copy_f32_kernel<<<cblocks256, 256, 0, stream>>>(bufC, bufA, (size_t)N * 256 / 4);
  scatter_add_kernel<<<eblocks, 256, 0, stream>>>(bufC, bufA, src, dst, E, 256);
  gemm_bias_act_kernel<256, 256, true><<<gblocks, 256, 0, stream>>>(bufA, w2aT, b2a, bufB, N);
  gemm_bias_act_kernel<256, 128, true><<<gblocks, 256, 0, stream>>>(bufB, w2bT, b2b, bufC, N);

  // ---------------- Layer 3: agg = h + scatter(h); MLP 128->128->128 --------
  copy_f32_kernel<<<cblocks128, 256, 0, stream>>>(bufC, bufA, (size_t)N * 128 / 4);
  scatter_add_kernel<<<eblocks, 256, 0, stream>>>(bufC, bufA, src, dst, E, 128);
  gemm_bias_act_kernel<128, 128, true><<<gblocks, 256, 0, stream>>>(bufA, w3aT, b3a, bufB, N);
  gemm_bias_act_kernel<128, 128, false><<<gblocks, 256, 0, stream>>>(bufB, w3bT, b3b,
                                                                     (float*)d_out, N);
}
